// DeepseekV3MoE_62989990363214
// MI455X (gfx1250) — matrix operand order repaired and checked by reading, not compiled, not run on hardware
//
#include <hip/hip_runtime.h>
#include <hip/hip_bf16.h>

// ---------------- problem constants ----------------
namespace {
constexpr int TNUM = 4096;   // tokens (B*S)
constexpr int HDIM = 1024;
constexpr int IDIM = 512;
constexpr int ENUM = 16;
constexpr int KSEL = 4;
constexpr int ROWT = 32;                                // GEMM row-tile (per block)
constexpr int SLOT_CAP = TNUM * KSEL + ENUM * ROWT;     // 16896 padded slots

// workspace layout (bytes), all 64B-aligned
constexpr size_t OFF_CNT   = 0;                                   // 16 ints
constexpr size_t OFF_POFF  = 256;                                 // 17 ints
constexpr size_t OFF_CUR   = 512;                                 // 16 ints
constexpr size_t OFF_TOPKI = 1024;                                // T*4 ints
constexpr size_t OFF_TOPKW = OFF_TOPKI + (size_t)TNUM * 4 * 4;    // T*4 floats
constexpr size_t OFF_TOK   = OFF_TOPKW + (size_t)TNUM * 4 * 4;    // SLOT_CAP ints
constexpr size_t OFF_WSL   = OFF_TOK + (size_t)SLOT_CAP * 4;      // SLOT_CAP floats
constexpr size_t OFF_XB    = OFF_WSL + (size_t)SLOT_CAP * 4 + 192;// x bf16 [T][H] (64B align)
constexpr size_t OFF_WG    = OFF_XB + (size_t)TNUM * HDIM * 2;    // wg^T bf16 [E][I][H]
constexpr size_t OFF_WU    = OFF_WG + (size_t)ENUM * IDIM * HDIM * 2;
constexpr size_t OFF_WD    = OFF_WU + (size_t)ENUM * IDIM * HDIM * 2; // wd^T [E][H][I]
constexpr size_t OFF_SG    = OFF_WD + (size_t)ENUM * HDIM * IDIM * 2; // wsg^T [I][H]
constexpr size_t OFF_SU    = OFF_SG + (size_t)HDIM * IDIM * 2;
constexpr size_t OFF_SD    = OFF_SU + (size_t)HDIM * IDIM * 2;        // wsd^T [H][I]
constexpr size_t OFF_INTS  = OFF_SD + (size_t)HDIM * IDIM * 2;        // inter shared bf16 [T][I]
constexpr size_t OFF_INTR  = OFF_INTS + (size_t)TNUM * IDIM * 2;      // inter routed bf16 [SLOT_CAP][I]

constexpr int AT_STRIDE = 40;   // LDS A-tile row stride in ushorts (80B: 16B aligned, bank-staggered)
} // namespace

// ---------------- WMMA types / helpers ----------------
typedef __attribute__((ext_vector_type(16))) __bf16 bf16x16;
typedef __attribute__((ext_vector_type(8)))  float  f32x8;

union FragU { bf16x16 v; uint4 q[2]; };

__device__ __forceinline__ unsigned short f2bf(float f) {
    unsigned int u = __float_as_uint(f);
    unsigned int r = u + 0x7FFFu + ((u >> 16) & 1u);   // RNE
    return (unsigned short)(r >> 16);
}

__device__ __forceinline__ f32x8 wmma_bf16(bf16x16 a, bf16x16 b, f32x8 c) {
    return __builtin_amdgcn_wmma_f32_16x16x32_bf16(false, a, false, b, (short)0, c, false, false);
}

// A fragment from an LDS-resident 32-wide k-chunk (rowPtr -> this lane's M row):
// lanes 0-15 K={0..7,16..23}; lanes 16-31 K={8..15,24..31}
__device__ __forceinline__ bf16x16 load_a_frag_lds(const unsigned short* rowPtr, int half) {
    FragU f;
    f.q[0] = *(const uint4*)(rowPtr + half * 8);
    f.q[1] = *(const uint4*)(rowPtr + 16 + half * 8);
    return f.v;
}

// B fragment (32x16 bf16), B stored [N][K]:
// lanes 0-15 N=lane K=k0+{0..7,16..23}; lanes 16-31 N=lane-16 K=k0+{8..15,24..31}
__device__ __forceinline__ bf16x16 load_b_frag(const unsigned short* __restrict__ colPtr,
                                               int k0, int lane) {
    FragU f;
    int half = lane >> 4;
    const unsigned short* p = colPtr + k0 + half * 8;
    f.q[0] = *(const uint4*)(p);
    f.q[1] = *(const uint4*)(p + 16);
    return f.v;
}

// one wave-wide async copy: each lane moves 16B global->LDS (ASYNCcnt-tracked)
__device__ __forceinline__ void async_copy16(unsigned lds_off, const unsigned short* gptr) {
    unsigned long long ga = (unsigned long long)(size_t)gptr;
    asm volatile("global_load_async_to_lds_b128 %0, %1, off"
                 :: "v"(lds_off), "v"(ga) : "memory");
}
__device__ __forceinline__ void async_wait(int pending) {
    if (pending) asm volatile("s_wait_asynccnt 0x1" ::: "memory");
    else         asm volatile("s_wait_asynccnt 0x0" ::: "memory");
}

// ---------------- router: one wave32 per token ----------------
__global__ __launch_bounds__(256) void moe_router(
    const float* __restrict__ x, const float* __restrict__ rw, const float* __restrict__ rb,
    int* __restrict__ topkIdx, float* __restrict__ topkW, int* __restrict__ counts,
    float* __restrict__ logitsOut)
{
    int lane = threadIdx.x & 31;
    int wid  = threadIdx.x >> 5;
    int t = blockIdx.x * 8 + wid;
    if (t >= TNUM) return;
    const float* xr = x + (size_t)t * HDIM;

    float acc[ENUM];
#pragma unroll
    for (int e = 0; e < ENUM; ++e) acc[e] = 0.f;
    for (int h = lane; h < HDIM; h += 32) {
        float xv = xr[h];
#pragma unroll
        for (int e = 0; e < ENUM; ++e) acc[e] += xv * rw[e * HDIM + h];
    }
#pragma unroll
    for (int e = 0; e < ENUM; ++e)
        for (int off = 16; off > 0; off >>= 1) acc[e] += __shfl_xor(acc[e], off, 32);

    float score[ENUM], sfc[ENUM];
#pragma unroll
    for (int e = 0; e < ENUM; ++e) {
        float s = 1.f / (1.f + __expf(-acc[e]));
        score[e] = s;
        sfc[e] = s + rb[e];
    }
    float gsc[4];
#pragma unroll
    for (int g = 0; g < 4; ++g) {
        float m1 = -1e30f, m2 = -1e30f;
#pragma unroll
        for (int j = 0; j < 4; ++j) {
            float v = sfc[g * 4 + j];
            if (v > m1) { m2 = m1; m1 = v; } else if (v > m2) m2 = v;
        }
        gsc[g] = m1 + m2;
    }
    int g1 = 0;
    for (int g = 1; g < 4; ++g) if (gsc[g] > gsc[g1]) g1 = g;
    int g2 = -1;
    for (int g = 0; g < 4; ++g) { if (g == g1) continue; if (g2 < 0 || gsc[g] > gsc[g2]) g2 = g; }

    float masked[ENUM];
#pragma unroll
    for (int e = 0; e < ENUM; ++e) {
        int g = e >> 2;
        masked[e] = (g == g1 || g == g2) ? sfc[e] : -1e30f;
    }
    int idx[KSEL]; float wsel[KSEL]; float wsum = 0.f;
    for (int k = 0; k < KSEL; ++k) {
        int best = 0;
        for (int e = 1; e < ENUM; ++e) if (masked[e] > masked[best]) best = e;
        idx[k] = best; wsel[k] = score[best]; wsum += score[best];
        masked[best] = -1e30f;
    }
    float nrm = 2.5f / (wsum + 1e-20f);
    if (lane == 0) {
        for (int k = 0; k < KSEL; ++k) {
            topkIdx[t * 4 + k] = idx[k];
            topkW[t * 4 + k]   = wsel[k] * nrm;
            atomicAdd(&counts[idx[k]], 1);
        }
        for (int e = 0; e < ENUM; ++e) logitsOut[(size_t)t * ENUM + e] = acc[e];
    }
}

// ---------------- offsets (32-row-aligned compaction) ----------------
__global__ void moe_offsets(const int* __restrict__ counts, int* __restrict__ poff,
                            int* __restrict__ cur) {
    if (threadIdx.x == 0 && blockIdx.x == 0) {
        int run = 0;
        for (int e = 0; e < ENUM; ++e) {
            poff[e] = run; cur[e] = run;
            run += ((counts[e] + ROWT - 1) / ROWT) * ROWT;
        }
        poff[ENUM] = run;
    }
}

__global__ __launch_bounds__(256) void moe_scatter(
    const int* __restrict__ topkIdx, const float* __restrict__ topkW,
    int* __restrict__ cur, int* __restrict__ tokOf, float* __restrict__ wOf)
{
    int id = blockIdx.x * 256 + threadIdx.x;
    if (id >= TNUM * KSEL) return;
    int e = topkIdx[id];
    int pos = atomicAdd(&cur[e], 1);
    tokOf[pos] = id >> 2;
    wOf[pos]   = topkW[id];
}

// ---------------- fp32 -> bf16 conversions ----------------
__global__ __launch_bounds__(256) void cvt_bf16(const float* __restrict__ src,
                                                unsigned short* __restrict__ dst, int n) {
    int id = blockIdx.x * 256 + threadIdx.x;
    if (id < n) dst[id] = f2bf(src[id]);
}

// batched transpose+convert: src f32 [b][R][C] -> dst bf16 [b][C][R]; R,C multiples of 32
__global__ __launch_bounds__(256) void transpose_cvt(const float* __restrict__ src,
                                                     unsigned short* __restrict__ dst,
                                                     int R, int C) {
    __shared__ unsigned short tile[32][33];
    size_t bo = (size_t)blockIdx.z * R * C;
    int c0 = blockIdx.x * 32, r0 = blockIdx.y * 32;
#pragma unroll
    for (int i = threadIdx.y; i < 32; i += 8)
        tile[i][threadIdx.x] = f2bf(src[bo + (size_t)(r0 + i) * C + c0 + threadIdx.x]);
    __syncthreads();
#pragma unroll
    for (int i = threadIdx.y; i < 32; i += 8)
        dst[bo + (size_t)(c0 + i) * R + r0 + threadIdx.x] = tile[threadIdx.x][i];
}

// ---------------- shared expert: gate/up + silu (32 rows x 64 cols / block) ------
__global__ __launch_bounds__(128) void shared_gateup(
    const unsigned short* __restrict__ xb, const unsigned short* __restrict__ wgT,
    const unsigned short* __restrict__ wuT, unsigned short* __restrict__ interS)
{
    __shared__ unsigned short atile[2][ROWT * AT_STRIDE];
    int tid = threadIdx.x;
    int lane = tid & 31, wv = tid >> 5;
    int m = lane & 15, half = lane >> 4;
    int row0 = blockIdx.x * ROWT;
    int col0 = blockIdx.y * 64 + wv * 16;

    int sr = tid >> 2, sc = (tid & 3) * 8;               // staging: 16B per lane
    const unsigned short* sgp = xb + (size_t)(row0 + sr) * HDIM + sc;
    unsigned ldsOff[2] = { (unsigned)(size_t)&atile[0][sr * AT_STRIDE + sc],
                           (unsigned)(size_t)&atile[1][sr * AT_STRIDE + sc] };
    const unsigned short* bgp = wgT + (size_t)(col0 + m) * HDIM;
    const unsigned short* bup = wuT + (size_t)(col0 + m) * HDIM;

    f32x8 ag0 = {}, ag1 = {}, au0 = {}, au1 = {};
    async_copy16(ldsOff[0], sgp);
    for (int kt = 0; kt < HDIM / 32; ++kt) {
        int k = kt * 32;
        bool more = (kt + 1) < HDIM / 32;
        if (more) async_copy16(ldsOff[(kt + 1) & 1], sgp + k + 32);
        async_wait(more);
        __syncthreads();
        const unsigned short* at = &atile[kt & 1][0];
        bf16x16 a0 = load_a_frag_lds(at + m * AT_STRIDE, half);
        bf16x16 a1 = load_a_frag_lds(at + (16 + m) * AT_STRIDE, half);
        bf16x16 fg = load_b_frag(bgp, k, lane);
        bf16x16 fu = load_b_frag(bup, k, lane);
        ag0 = wmma_bf16(a0, fg, ag0);
        ag1 = wmma_bf16(a1, fg, ag1);
        au0 = wmma_bf16(a0, fu, au0);
        au1 = wmma_bf16(a1, fu, au1);
        __syncthreads();
    }
#pragma unroll
    for (int j = 0; j < 8; ++j) {
        float g0 = ag0[j], u0 = au0[j];
        float g1 = ag1[j], u1 = au1[j];
        float s0 = (g0 / (1.f + __expf(-g0))) * u0;
        float s1 = (g1 / (1.f + __expf(-g1))) * u1;
        interS[(size_t)(row0 + j + half * 8) * IDIM + col0 + m]      = f2bf(s0);
        interS[(size_t)(row0 + 16 + j + half * 8) * IDIM + col0 + m] = f2bf(s1);
    }
}

// ---------------- shared expert: down (32 rows x 128 cols / block) ----------------
__global__ __launch_bounds__(128) void shared_down(
    const unsigned short* __restrict__ interS, const unsigned short* __restrict__ sdT,
    float* __restrict__ out)
{
    __shared__ unsigned short atile[2][ROWT * AT_STRIDE];
    int tid = threadIdx.x;
    int lane = tid & 31, wv = tid >> 5;
    int m = lane & 15, half = lane >> 4;
    int row0 = blockIdx.x * ROWT;
    int col0 = blockIdx.y * 128 + wv * 32;

    int sr = tid >> 2, sc = (tid & 3) * 8;
    const unsigned short* sgp = interS + (size_t)(row0 + sr) * IDIM + sc;
    unsigned ldsOff[2] = { (unsigned)(size_t)&atile[0][sr * AT_STRIDE + sc],
                           (unsigned)(size_t)&atile[1][sr * AT_STRIDE + sc] };
    const unsigned short* b0 = sdT + (size_t)(col0 + m) * IDIM;
    const unsigned short* b1 = sdT + (size_t)(col0 + 16 + m) * IDIM;

    f32x8 c00 = {}, c01 = {}, c10 = {}, c11 = {};
    async_copy16(ldsOff[0], sgp);
    for (int kt = 0; kt < IDIM / 32; ++kt) {
        int k = kt * 32;
        bool more = (kt + 1) < IDIM / 32;
        if (more) async_copy16(ldsOff[(kt + 1) & 1], sgp + k + 32);
        async_wait(more);
        __syncthreads();
        const unsigned short* at = &atile[kt & 1][0];
        bf16x16 a0 = load_a_frag_lds(at + m * AT_STRIDE, half);
        bf16x16 a1 = load_a_frag_lds(at + (16 + m) * AT_STRIDE, half);
        bf16x16 f0 = load_b_frag(b0, k, lane);
        bf16x16 f1 = load_b_frag(b1, k, lane);
        c00 = wmma_bf16(a0, f0, c00);
        c01 = wmma_bf16(a0, f1, c01);
        c10 = wmma_bf16(a1, f0, c10);
        c11 = wmma_bf16(a1, f1, c11);
        __syncthreads();
    }
#pragma unroll
    for (int j = 0; j < 8; ++j) {
        int r0 = row0 + j + half * 8, r1 = r0 + 16;
        out[(size_t)r0 * HDIM + col0 + m]      = c00[j];
        out[(size_t)r0 * HDIM + col0 + 16 + m] = c01[j];
        out[(size_t)r1 * HDIM + col0 + m]      = c10[j];
        out[(size_t)r1 * HDIM + col0 + 16 + m] = c11[j];
    }
}

// ---------------- routed experts: gate/up + silu (token gather at staging) -------
__global__ __launch_bounds__(128) void routed_gateup(
    const unsigned short* __restrict__ xb, const unsigned short* __restrict__ wgT,
    const unsigned short* __restrict__ wuT, const int* __restrict__ counts,
    const int* __restrict__ poff, const int* __restrict__ tokOf,
    unsigned short* __restrict__ interR)
{
    __shared__ unsigned short atile[2][ROWT * AT_STRIDE];
    int e = blockIdx.z;
    int cnt = counts[e];
    int rt = blockIdx.x;
    if (rt * ROWT >= cnt) return;
    int base = poff[e] + rt * ROWT;
    int tid = threadIdx.x;
    int lane = tid & 31, wv = tid >> 5;
    int m = lane & 15, half = lane >> 4;
    int col0 = blockIdx.y * 64 + wv * 16;

    int sr = tid >> 2, sc = (tid & 3) * 8;
    int stok = tokOf[base + sr];                        // padded slots -> token 0
    const unsigned short* sgp = xb + (size_t)stok * HDIM + sc;
    unsigned ldsOff[2] = { (unsigned)(size_t)&atile[0][sr * AT_STRIDE + sc],
                           (unsigned)(size_t)&atile[1][sr * AT_STRIDE + sc] };
    const unsigned short* bgp = wgT + ((size_t)e * IDIM + col0 + m) * HDIM;
    const unsigned short* bup = wuT + ((size_t)e * IDIM + col0 + m) * HDIM;

    f32x8 ag0 = {}, ag1 = {}, au0 = {}, au1 = {};
    async_copy16(ldsOff[0], sgp);
    for (int kt = 0; kt < HDIM / 32; ++kt) {
        int k = kt * 32;
        bool more = (kt + 1) < HDIM / 32;
        if (more) async_copy16(ldsOff[(kt + 1) & 1], sgp + k + 32);
        async_wait(more);
        __syncthreads();
        const unsigned short* at = &atile[kt & 1][0];
        bf16x16 a0 = load_a_frag_lds(at + m * AT_STRIDE, half);
        bf16x16 a1 = load_a_frag_lds(at + (16 + m) * AT_STRIDE, half);
        bf16x16 fg = load_b_frag(bgp, k, lane);
        bf16x16 fu = load_b_frag(bup, k, lane);
        ag0 = wmma_bf16(a0, fg, ag0);
        ag1 = wmma_bf16(a1, fg, ag1);
        au0 = wmma_bf16(a0, fu, au0);
        au1 = wmma_bf16(a1, fu, au1);
        __syncthreads();
    }
#pragma unroll
    for (int j = 0; j < 8; ++j) {
        float g0 = ag0[j], u0 = au0[j];
        float g1 = ag1[j], u1 = au1[j];
        float s0 = (g0 / (1.f + __expf(-g0))) * u0;
        float s1 = (g1 / (1.f + __expf(-g1))) * u1;
        interR[(size_t)(base + j + half * 8) * IDIM + col0 + m]      = f2bf(s0);
        interR[(size_t)(base + 16 + j + half * 8) * IDIM + col0 + m] = f2bf(s1);
    }
}

// ---------------- routed experts: down + weighted atomic combine ----------------
__global__ __launch_bounds__(128) void routed_down(
    const unsigned short* __restrict__ interR, const unsigned short* __restrict__ wdT,
    const int* __restrict__ counts, const int* __restrict__ poff,
    const int* __restrict__ tokOf, const float* __restrict__ wOf,
    float* __restrict__ out)
{
    __shared__ unsigned short atile[2][ROWT * AT_STRIDE];
    int e = blockIdx.z;
    int cnt = counts[e];
    int rt = blockIdx.x;
    if (rt * ROWT >= cnt) return;
    int base = poff[e] + rt * ROWT;
    int tid = threadIdx.x;
    int lane = tid & 31, wv = tid >> 5;
    int m = lane & 15, half = lane >> 4;
    int col0 = blockIdx.y * 128 + wv * 32;

    int sr = tid >> 2, sc = (tid & 3) * 8;
    const unsigned short* sgp = interR + (size_t)(base + sr) * IDIM + sc;
    unsigned ldsOff[2] = { (unsigned)(size_t)&atile[0][sr * AT_STRIDE + sc],
                           (unsigned)(size_t)&atile[1][sr * AT_STRIDE + sc] };
    const unsigned short* b0 = wdT + ((size_t)e * HDIM + col0 + m) * IDIM;
    const unsigned short* b1 = wdT + ((size_t)e * HDIM + col0 + 16 + m) * IDIM;

    f32x8 c00 = {}, c01 = {}, c10 = {}, c11 = {};
    async_copy16(ldsOff[0], sgp);
    for (int kt = 0; kt < IDIM / 32; ++kt) {
        int k = kt * 32;
        bool more = (kt + 1) < IDIM / 32;
        if (more) async_copy16(ldsOff[(kt + 1) & 1], sgp + k + 32);
        async_wait(more);
        __syncthreads();
        const unsigned short* at = &atile[kt & 1][0];
        bf16x16 a0 = load_a_frag_lds(at + m * AT_STRIDE, half);
        bf16x16 a1 = load_a_frag_lds(at + (16 + m) * AT_STRIDE, half);
        bf16x16 f0 = load_b_frag(b0, k, lane);
        bf16x16 f1 = load_b_frag(b1, k, lane);
        c00 = wmma_bf16(a0, f0, c00);
        c01 = wmma_bf16(a0, f1, c01);
        c10 = wmma_bf16(a1, f0, c10);
        c11 = wmma_bf16(a1, f1, c11);
        __syncthreads();
    }
#pragma unroll
    for (int j = 0; j < 8; ++j) {
#pragma unroll
        for (int rr = 0; rr < 2; ++rr) {
            int r = rr * 16 + j + half * 8;
            if (rt * ROWT + r < cnt) {
                int slot = base + r;
                int tok  = tokOf[slot];
                float w  = wOf[slot];
                float v0 = (rr == 0) ? c00[j] : c10[j];
                float v1 = (rr == 0) ? c01[j] : c11[j];
                atomicAdd(&out[(size_t)tok * HDIM + col0 + m],      w * v0);
                atomicAdd(&out[(size_t)tok * HDIM + col0 + 16 + m], w * v1);
            }
        }
    }
}

// ---------------- host launcher ----------------
extern "C" void kernel_launch(void* const* d_in, const int* in_sizes, int n_in,
                              void* d_out, int out_size, void* d_ws, size_t ws_size,
                              hipStream_t stream) {
    (void)in_sizes; (void)n_in; (void)out_size; (void)ws_size;
    const float* x  = (const float*)d_in[0];
    const float* rw = (const float*)d_in[1];
    const float* rb = (const float*)d_in[2];
    const float* wg = (const float*)d_in[3];
    const float* wu = (const float*)d_in[4];
    const float* wd = (const float*)d_in[5];
    const float* sg = (const float*)d_in[6];
    const float* su = (const float*)d_in[7];
    const float* sd = (const float*)d_in[8];

    float* out       = (float*)d_out;
    float* logitsOut = out + (size_t)TNUM * HDIM;

    char* ws = (char*)d_ws;
    int*   counts  = (int*)(ws + OFF_CNT);
    int*   poff    = (int*)(ws + OFF_POFF);
    int*   cur     = (int*)(ws + OFF_CUR);
    int*   topkIdx = (int*)(ws + OFF_TOPKI);
    float* topkW   = (float*)(ws + OFF_TOPKW);
    int*   tokOf   = (int*)(ws + OFF_TOK);
    float* wOf     = (float*)(ws + OFF_WSL);
    unsigned short* xb     = (unsigned short*)(ws + OFF_XB);
    unsigned short* wgT    = (unsigned short*)(ws + OFF_WG);
    unsigned short* wuT    = (unsigned short*)(ws + OFF_WU);
    unsigned short* wdT    = (unsigned short*)(ws + OFF_WD);
    unsigned short* sgT    = (unsigned short*)(ws + OFF_SG);
    unsigned short* suT    = (unsigned short*)(ws + OFF_SU);
    unsigned short* sdT    = (unsigned short*)(ws + OFF_SD);
    unsigned short* interS = (unsigned short*)(ws + OFF_INTS);
    unsigned short* interR = (unsigned short*)(ws + OFF_INTR);

    // zero counters + slot arrays (tok/w contiguous => one memset)
    hipMemsetAsync(ws + OFF_CNT, 0, 64, stream);
    hipMemsetAsync(ws + OFF_TOK, 0, (size_t)SLOT_CAP * 4 * 2, stream);

    // routing
    moe_router<<<TNUM / 8, 256, 0, stream>>>(x, rw, rb, topkIdx, topkW, counts, logitsOut);
    moe_offsets<<<1, 32, 0, stream>>>(counts, poff, cur);
    moe_scatter<<<(TNUM * KSEL + 255) / 256, 256, 0, stream>>>(topkIdx, topkW, cur, tokOf, wOf);

    // bf16 conversions (x row-major; all weights transposed to [N][K])
    cvt_bf16<<<(TNUM * HDIM) / 256, 256, 0, stream>>>(x, xb, TNUM * HDIM);
    transpose_cvt<<<dim3(IDIM / 32, HDIM / 32, ENUM), dim3(32, 8), 0, stream>>>(wg, wgT, HDIM, IDIM);
    transpose_cvt<<<dim3(IDIM / 32, HDIM / 32, ENUM), dim3(32, 8), 0, stream>>>(wu, wuT, HDIM, IDIM);
    transpose_cvt<<<dim3(HDIM / 32, IDIM / 32, ENUM), dim3(32, 8), 0, stream>>>(wd, wdT, IDIM, HDIM);
    transpose_cvt<<<dim3(IDIM / 32, HDIM / 32, 1),    dim3(32, 8), 0, stream>>>(sg, sgT, HDIM, IDIM);
    transpose_cvt<<<dim3(IDIM / 32, HDIM / 32, 1),    dim3(32, 8), 0, stream>>>(su, suT, HDIM, IDIM);
    transpose_cvt<<<dim3(HDIM / 32, IDIM / 32, 1),    dim3(32, 8), 0, stream>>>(sd, sdT, IDIM, HDIM);

    // shared expert (down pass fully writes d_out, no atomics needed)
    shared_gateup<<<dim3(TNUM / ROWT, IDIM / 64), 128, 0, stream>>>(xb, sgT, suT, interS);
    shared_down  <<<dim3(TNUM / ROWT, HDIM / 128), 128, 0, stream>>>(interS, sdT, out);

    // routed experts (sparse top-4), combine via weighted atomicAdd into d_out
    routed_gateup<<<dim3(TNUM / ROWT, IDIM / 64, ENUM), 128, 0, stream>>>(
        xb, wgT, wuT, counts, poff, tokOf, interR);
    routed_down<<<dim3(TNUM / ROWT, HDIM / 128, ENUM), 128, 0, stream>>>(
        interR, wdT, counts, poff, tokOf, wOf, out);
}
